// GPT_4518305595537
// MI455X (gfx1250) — matrix operand order repaired and checked by reading, not compiled, not run on hardware
//
#include <hip/hip_runtime.h>
#include <math.h>

// ---------------- fixed problem dims ----------------
#define NLAY 8
#define CDIM 1024
#define HEADS 16
#define DHEAD 64
#define BB 8
#define TT 256
#define LREAL 511
#define LPAD 512
#define MALL (BB*LPAD)          // 4096
#define FF 4096
#define ADIM 8
#define EPSLN 1e-5f

typedef unsigned short u16;
typedef __attribute__((ext_vector_type(16))) __bf16  v16bf;
typedef __attribute__((ext_vector_type(8)))  float   v8f;
typedef __attribute__((ext_vector_type(8)))  u16     v8us;

union Frag { v16bf v; v8us h[2]; };

__device__ __forceinline__ u16 f2bf(float f) {
    unsigned u = __float_as_uint(f);
    unsigned r = (u + 0x7FFFu + ((u >> 16) & 1u)) >> 16;
    return (u16)r;
}

// A-fragment (16x32, 16-bit): lane holds K = lh*16 + 0..15 contiguous
__device__ __forceinline__ v16bf ld_frag_gap16(const u16* p) {
    Frag f; f.h[0] = *(const v8us*)p; f.h[1] = *(const v8us*)(p + 8); return f.v;
}
// B-fragment (32x16, 16-bit): lane holds K = lh*16 + 0..15 contiguous
__device__ __forceinline__ v16bf ld_frag_contig(const u16* p) {
    Frag f; f.h[0] = *(const v8us*)p; f.h[1] = *(const v8us*)(p + 8); return f.v;
}
__device__ __forceinline__ v16bf ld_frag_lds(const u16* p) {
    Frag f; f.h[0] = *(const v8us*)p; f.h[1] = *(const v8us*)(p + 8); return f.v;
}

__device__ __forceinline__ v8f wmma_bf16(v16bf a, v16bf b, v8f c) {
    return __builtin_amdgcn_wmma_f32_16x16x32_bf16(false, a, false, b, (short)0, c, false, false);
}

// ================= embed / interleave =================
__global__ __launch_bounds__(256)
void k_embed(const float* __restrict__ st, const float* __restrict__ act,
             const int* __restrict__ ts, const float* __restrict__ posem,
             const float* __restrict__ gpe, float* __restrict__ x) {
    int gm = blockIdx.x;                 // 0..4095
    int b = gm >> 9, l = gm & (LPAD - 1);
    int c0 = threadIdx.x * 4;
    float* xr = x + (size_t)gm * CDIM + c0;
    if (l >= LREAL) { xr[0]=0.f; xr[1]=0.f; xr[2]=0.f; xr[3]=0.f; return; }
    const float* tok = (l & 1)
        ? act + ((size_t)b * (TT - 1) + (l >> 1)) * CDIM
        : st  + ((size_t)b * TT       + (l >> 1)) * CDIM;
    const float* g = gpe + (size_t)ts[b] * CDIM;
    const float* p = posem + (size_t)l * CDIM;
#pragma unroll
    for (int i = 0; i < 4; ++i)
        xr[i] = tok[c0 + i] + g[c0 + i] + p[c0 + i];
}

// ================= layernorm -> bf16 =================
__global__ __launch_bounds__(256)
void k_ln(const float* __restrict__ x, const float* __restrict__ g,
          const float* __restrict__ be, u16* __restrict__ out) {
    __shared__ float red[2][8];
    __shared__ float bc[2];
    int gm = blockIdx.x;
    int l = gm & (LPAD - 1);
    int c0 = threadIdx.x * 4;
    u16* orow = out + (size_t)gm * CDIM + c0;
    if (l >= LREAL) { orow[0]=0; orow[1]=0; orow[2]=0; orow[3]=0; return; }
    const float* xr = x + (size_t)gm * CDIM + c0;
    float v[4]; float s = 0.f, sq = 0.f;
#pragma unroll
    for (int i = 0; i < 4; ++i) { v[i] = xr[i]; s += v[i]; sq += v[i]*v[i]; }
    for (int o = 16; o; o >>= 1) { s += __shfl_xor(s, o); sq += __shfl_xor(sq, o); }
    int lane = threadIdx.x & 31, wid = threadIdx.x >> 5;
    if (lane == 0) { red[0][wid] = s; red[1][wid] = sq; }
    __syncthreads();
    if (threadIdx.x == 0) {
        float ts = 0.f, tq = 0.f;
        for (int i = 0; i < 8; ++i) { ts += red[0][i]; tq += red[1][i]; }
        float mu = ts * (1.0f / CDIM);
        float var = tq * (1.0f / CDIM) - mu * mu;
        bc[0] = mu; bc[1] = rsqrtf(var + EPSLN);
    }
    __syncthreads();
    float mu = bc[0], rs = bc[1];
#pragma unroll
    for (int i = 0; i < 4; ++i)
        orow[i] = f2bf((v[i] - mu) * rs * g[c0 + i] + be[c0 + i]);
}

// ================= weight transpose+convert f32[K][N] -> bf16[N][K] =========
__global__ __launch_bounds__(256)
void k_wt(const float* __restrict__ in, u16* __restrict__ out, int K, int N) {
    __shared__ float t[32][33];
    int n0 = blockIdx.x * 32, k0 = blockIdx.y * 32;
    int tx = threadIdx.x, ty = threadIdx.y;
#pragma unroll
    for (int i = 0; i < 4; ++i)
        t[ty + i * 8][tx] = in[(size_t)(k0 + ty + i * 8) * N + n0 + tx];
    __syncthreads();
#pragma unroll
    for (int i = 0; i < 4; ++i)
        out[(size_t)(n0 + ty + i * 8) * K + k0 + tx] = f2bf(t[tx][ty + i * 8]);
}

// ================= WMMA GEMM: out[M=4096][N] = A[M][K] * Wt[N][K]^T + bias ==
// Block: 256 thr = 8 waves (4M x 2N). Block tile 128M x 128N.
// Wave tile 32M x 64N: 2 A-frags + 4 B-frags (12 b128 loads) -> 8 WMMA / K-step.
#define EP_RESID 0
#define EP_GELU  1
#define EP_QK    2
#define EP_VT    3

template<int MODE, int N, int K>
__global__ __launch_bounds__(256)
void k_gemm(const u16* __restrict__ A, const u16* __restrict__ Wt,
            const float* __restrict__ bias, float* __restrict__ xres,
            u16* __restrict__ obf) {
    int lane = threadIdx.x & 31, wid = threadIdx.x >> 5;
    int wm = wid & 3, wn = wid >> 2;
    int m0 = blockIdx.x * 128 + wm * 32;
    int n0 = blockIdx.y * 128 + wn * 64;
    int lr = lane & 15, lh = lane >> 4;

    v8f acc[2][4];
#pragma unroll
    for (int mi = 0; mi < 2; ++mi)
#pragma unroll
        for (int ni = 0; ni < 4; ++ni) acc[mi][ni] = (v8f){};

    const u16* ar0 = A + (size_t)(m0 + lr) * K + lh * 16;
    const u16* ar1 = ar0 + (size_t)16 * K;
    const u16* br0 = Wt + (size_t)(n0 + lr) * K + lh * 16;
    const u16* br1 = br0 + (size_t)16 * K;
    const u16* br2 = br0 + (size_t)32 * K;
    const u16* br3 = br0 + (size_t)48 * K;

#pragma unroll 2
    for (int kk = 0; kk < K; kk += 32) {
        v16bf a0 = ld_frag_gap16(ar0 + kk);
        v16bf a1 = ld_frag_gap16(ar1 + kk);
        v16bf b0 = ld_frag_contig(br0 + kk);
        v16bf b1 = ld_frag_contig(br1 + kk);
        v16bf b2 = ld_frag_contig(br2 + kk);
        v16bf b3 = ld_frag_contig(br3 + kk);
        acc[0][0] = wmma_bf16(a0, b0, acc[0][0]);
        acc[0][1] = wmma_bf16(a0, b1, acc[0][1]);
        acc[0][2] = wmma_bf16(a0, b2, acc[0][2]);
        acc[0][3] = wmma_bf16(a0, b3, acc[0][3]);
        acc[1][0] = wmma_bf16(a1, b0, acc[1][0]);
        acc[1][1] = wmma_bf16(a1, b1, acc[1][1]);
        acc[1][2] = wmma_bf16(a1, b2, acc[1][2]);
        acc[1][3] = wmma_bf16(a1, b3, acc[1][3]);
    }

#pragma unroll
    for (int mi = 0; mi < 2; ++mi) {
#pragma unroll
        for (int ni = 0; ni < 4; ++ni) {
            v8f av = acc[mi][ni];
            int mt = m0 + mi * 16, nt = n0 + ni * 16;
#pragma unroll
            for (int r = 0; r < 8; ++r) {
                int gm = mt + r + 8 * lh;
                int gn = nt + lr;
                float v = av[r] + bias[gn];
                if (MODE == EP_RESID) {
                    size_t idx = (size_t)gm * N + gn;
                    xres[idx] = xres[idx] + v;
                } else if (MODE == EP_GELU) {
                    float gl = 0.5f * v * (1.0f + erff(v * 0.70710678118654752f));
                    obf[(size_t)gm * N + gn] = f2bf(gl);
                } else if (MODE == EP_QK) {
                    int b = gm >> 9, l = gm & 511, h = gn >> 6, d = gn & 63;
                    obf[(size_t)(b * HEADS + h) * (LPAD * DHEAD) + l * DHEAD + d] = f2bf(v);
                } else { // EP_VT
                    int b = gm >> 9, l = gm & 511, h = gn >> 6, d = gn & 63;
                    obf[(size_t)(b * HEADS + h) * (LPAD * DHEAD) + d * LPAD + l] = f2bf(v);
                }
            }
        }
    }
}

// ================= flash attention (per (b,h,qtile) wave) ====================
__global__ __launch_bounds__(128)
void k_attn(const u16* __restrict__ q, const u16* __restrict__ kbuf,
            const u16* __restrict__ vt, u16* __restrict__ y) {
    __shared__ __align__(16) u16 sP[4][16][48];
    int lane = threadIdx.x & 31, wid = threadIdx.x >> 5;
    int gw = blockIdx.x * 4 + wid;
    int qt = gw & 31;            // LPAD/16 = 32 q-tiles
    int bh = gw >> 5;            // b*HEADS + h
    int lr = lane & 15, lh = lane >> 4;
    int m0 = qt * 16;

    const u16* Q  = q    + (size_t)bh * (LPAD * DHEAD);
    const u16* Kp = kbuf + (size_t)bh * (LPAD * DHEAD);
    const u16* Vt = vt   + (size_t)bh * (LPAD * DHEAD);

    v16bf aq0 = ld_frag_gap16(Q + (size_t)(m0 + lr) * DHEAD + 0  + lh * 16);
    v16bf aq1 = ld_frag_gap16(Q + (size_t)(m0 + lr) * DHEAD + 32 + lh * 16);

    float mrow[8], lrow[8];
    v8f o0 = {}, o1 = {}, o2 = {}, o3 = {};
#pragma unroll
    for (int r = 0; r < 8; ++r) { mrow[r] = -1e30f; lrow[r] = 0.f; }

    int nblocks = (m0 + 16 + 31) >> 5;
    for (int kb = 0; kb < nblocks; ++kb) {
        int n0 = kb * 32;
        v16bf bk0a = ld_frag_contig(Kp + (size_t)(n0 + lr) * DHEAD + 0  + lh * 16);
        v16bf bk0b = ld_frag_contig(Kp + (size_t)(n0 + lr) * DHEAD + 32 + lh * 16);
        v16bf bk1a = ld_frag_contig(Kp + (size_t)(n0 + 16 + lr) * DHEAD + 0  + lh * 16);
        v16bf bk1b = ld_frag_contig(Kp + (size_t)(n0 + 16 + lr) * DHEAD + 32 + lh * 16);
        v8f s0 = {}, s1 = {};
        s0 = wmma_bf16(aq0, bk0a, s0); s0 = wmma_bf16(aq1, bk0b, s0);
        s1 = wmma_bf16(aq0, bk1a, s1); s1 = wmma_bf16(aq1, bk1b, s1);

        float alp[8];
#pragma unroll
        for (int r = 0; r < 8; ++r) {
            int qrow = m0 + r + 8 * lh;
            float v0 = s0[r] * 0.125f; if (n0 + lr > qrow)       v0 = -1e30f;
            float v1 = s1[r] * 0.125f; if (n0 + 16 + lr > qrow)  v1 = -1e30f;
            float mx = fmaxf(v0, v1);
            for (int o = 1; o < 16; o <<= 1) mx = fmaxf(mx, __shfl_xor(mx, o));
            float mn = fmaxf(mrow[r], mx);
            float alpha = __expf(mrow[r] - mn);
            float p0 = __expf(v0 - mn), p1 = __expf(v1 - mn);
            float rs = p0 + p1;
            for (int o = 1; o < 16; o <<= 1) rs += __shfl_xor(rs, o);
            lrow[r] = lrow[r] * alpha + rs;
            mrow[r] = mn;
            alp[r] = alpha;
            sP[wid][r + 8 * lh][lr]      = f2bf(p0);
            sP[wid][r + 8 * lh][16 + lr] = f2bf(p1);
        }
#pragma unroll
        for (int r = 0; r < 8; ++r) {
            o0[r] *= alp[r]; o1[r] *= alp[r]; o2[r] *= alp[r]; o3[r] *= alp[r];
        }
        v16bf ap = ld_frag_lds(&sP[wid][lr][lh * 16]);
        v16bf bv0 = ld_frag_contig(Vt + (size_t)(0  + lr) * LPAD + n0 + lh * 16);
        v16bf bv1 = ld_frag_contig(Vt + (size_t)(16 + lr) * LPAD + n0 + lh * 16);
        v16bf bv2 = ld_frag_contig(Vt + (size_t)(32 + lr) * LPAD + n0 + lh * 16);
        v16bf bv3 = ld_frag_contig(Vt + (size_t)(48 + lr) * LPAD + n0 + lh * 16);
        o0 = wmma_bf16(ap, bv0, o0);
        o1 = wmma_bf16(ap, bv1, o1);
        o2 = wmma_bf16(ap, bv2, o2);
        o3 = wmma_bf16(ap, bv3, o3);
    }

    float rinv[8];
#pragma unroll
    for (int r = 0; r < 8; ++r) rinv[r] = lrow[r] > 0.f ? 1.0f / lrow[r] : 0.f;
    int b = bh >> 4, h = bh & 15;
    u16* yb = y + ((size_t)b * LPAD + m0) * CDIM + h * DHEAD;
#pragma unroll
    for (int r = 0; r < 8; ++r) {
        size_t row = (size_t)(r + 8 * lh) * CDIM;
        yb[row + 0  + lr] = f2bf(o0[r] * rinv[r]);
        yb[row + 16 + lr] = f2bf(o1[r] * rinv[r]);
        yb[row + 32 + lr] = f2bf(o2[r] * rinv[r]);
        yb[row + 48 + lr] = f2bf(o3[r] * rinv[r]);
    }
}

// ================= final LN + head, state rows only ==========================
__global__ __launch_bounds__(256)
void k_head(const float* __restrict__ x, const float* __restrict__ g,
            const float* __restrict__ be, const float* __restrict__ hw,
            float* __restrict__ out) {
    __shared__ float red[2][8];
    __shared__ float bc[2];
    __shared__ float ared[8][ADIM];
    int bt = blockIdx.x;
    int b = bt >> 8, t = bt & 255;
    int gm = b * LPAD + 2 * t;
    int c0 = threadIdx.x * 4;
    const float* xr = x + (size_t)gm * CDIM + c0;
    float v[4]; float s = 0.f, sq = 0.f;
#pragma unroll
    for (int i = 0; i < 4; ++i) { v[i] = xr[i]; s += v[i]; sq += v[i]*v[i]; }
    for (int o = 16; o; o >>= 1) { s += __shfl_xor(s, o); sq += __shfl_xor(sq, o); }
    int lane = threadIdx.x & 31, wid = threadIdx.x >> 5;
    if (lane == 0) { red[0][wid] = s; red[1][wid] = sq; }
    __syncthreads();
    if (threadIdx.x == 0) {
        float ts = 0.f, tq = 0.f;
        for (int i = 0; i < 8; ++i) { ts += red[0][i]; tq += red[1][i]; }
        float mu = ts * (1.0f / CDIM);
        bc[0] = mu; bc[1] = rsqrtf(tq * (1.0f / CDIM) - mu * mu + EPSLN);
    }
    __syncthreads();
    float mu = bc[0], rs = bc[1];
    float acc[ADIM];
#pragma unroll
    for (int a = 0; a < ADIM; ++a) acc[a] = 0.f;
#pragma unroll
    for (int i = 0; i < 4; ++i) {
        float xn = (v[i] - mu) * rs * g[c0 + i] + be[c0 + i];
        const float* hr = hw + (size_t)(c0 + i) * ADIM;
#pragma unroll
        for (int a = 0; a < ADIM; ++a) acc[a] += xn * hr[a];
    }
#pragma unroll
    for (int a = 0; a < ADIM; ++a)
        for (int o = 16; o; o >>= 1) acc[a] += __shfl_xor(acc[a], o);
    if (lane == 0)
#pragma unroll
        for (int a = 0; a < ADIM; ++a) ared[wid][a] = acc[a];
    __syncthreads();
    if (threadIdx.x < ADIM) {
        float tot = 0.f;
        for (int w = 0; w < 8; ++w) tot += ared[w][threadIdx.x];
        out[(size_t)bt * ADIM + threadIdx.x] = tot;
    }
}

// =============================== host ========================================
extern "C" void kernel_launch(void* const* d_in, const int* in_sizes, int n_in,
                              void* d_out, int out_size, void* d_ws, size_t ws_size,
                              hipStream_t stream) {
    const float* st    = (const float*)d_in[0];
    const float* act   = (const float*)d_in[1];
    const int*   tsp   = (const int*)  d_in[2];
    const float* posem = (const float*)d_in[3];
    const float* gpe   = (const float*)d_in[4];
    const float* ln1g  = (const float*)d_in[5];
    const float* ln1b  = (const float*)d_in[6];
    const float* wq    = (const float*)d_in[7];
    const float* bq    = (const float*)d_in[8];
    const float* wk    = (const float*)d_in[9];
    const float* bk    = (const float*)d_in[10];
    const float* wv    = (const float*)d_in[11];
    const float* bv    = (const float*)d_in[12];
    const float* wp    = (const float*)d_in[13];
    const float* bp    = (const float*)d_in[14];
    const float* ln2g  = (const float*)d_in[15];
    const float* ln2b  = (const float*)d_in[16];
    const float* w1    = (const float*)d_in[17];
    const float* b1    = (const float*)d_in[18];
    const float* w2    = (const float*)d_in[19];
    const float* b2    = (const float*)d_in[20];
    const float* lnfg  = (const float*)d_in[21];
    const float* lnfb  = (const float*)d_in[22];
    const float* headw = (const float*)d_in[23];
    float* out = (float*)d_out;

    char* w = (char*)d_ws;
    float* x = (float*)w;            w += (size_t)MALL * CDIM * 4;   // 16 MB
    u16* h   = (u16*)w;              w += (size_t)MALL * CDIM * 2;   // 8 MB
    u16* R   = (u16*)w;              w += (size_t)MALL * FF * 2;     // 32 MB (aliased)
    u16* qb  = R;
    u16* kb  = qb + (size_t)4194304;
    u16* vtb = kb + (size_t)4194304;
    u16* yb  = vtb + (size_t)4194304;
    u16* hid = R;                                   // aliases q/k/vt/y (dead then)
    u16* wtq = (u16*)w;              w += (size_t)CDIM * CDIM * 2;
    u16* wtk = (u16*)w;              w += (size_t)CDIM * CDIM * 2;
    u16* wtv = (u16*)w;              w += (size_t)CDIM * CDIM * 2;
    u16* wtp = (u16*)w;              w += (size_t)CDIM * CDIM * 2;
    u16* wt1 = (u16*)w;              w += (size_t)CDIM * FF * 2;
    u16* wt2 = (u16*)w;              w += (size_t)FF * CDIM * 2;
    (void)ws_size; (void)in_sizes; (void)n_in; (void)out_size;

    dim3 tb256(256), tb128(128);
    dim3 gN1(32, 8);    // N=1024: block tile 128x128
    dim3 gN4(32, 32);   // N=4096
    dim3 twt(32, 8);

    k_embed<<<MALL, tb256, 0, stream>>>(st, act, tsp, posem, gpe, x);

    for (int ly = 0; ly < NLAY; ++ly) {
        size_t wcc = (size_t)ly * CDIM * CDIM;
        size_t wcf = (size_t)ly * CDIM * FF;
        k_wt<<<dim3(CDIM/32, CDIM/32), twt, 0, stream>>>(wq + wcc, wtq, CDIM, CDIM);
        k_wt<<<dim3(CDIM/32, CDIM/32), twt, 0, stream>>>(wk + wcc, wtk, CDIM, CDIM);
        k_wt<<<dim3(CDIM/32, CDIM/32), twt, 0, stream>>>(wv + wcc, wtv, CDIM, CDIM);
        k_wt<<<dim3(CDIM/32, CDIM/32), twt, 0, stream>>>(wp + wcc, wtp, CDIM, CDIM);
        k_wt<<<dim3(FF/32,  CDIM/32), twt, 0, stream>>>(w1 + wcf, wt1, CDIM, FF);
        k_wt<<<dim3(CDIM/32, FF/32),  twt, 0, stream>>>(w2 + wcf, wt2, FF, CDIM);

        k_ln<<<MALL, tb256, 0, stream>>>(x, ln1g + (size_t)ly*CDIM, ln1b + (size_t)ly*CDIM, h);

        k_gemm<EP_QK, CDIM, CDIM><<<gN1, tb256, 0, stream>>>(h, wtq, bq + (size_t)ly*CDIM, nullptr, qb);
        k_gemm<EP_QK, CDIM, CDIM><<<gN1, tb256, 0, stream>>>(h, wtk, bk + (size_t)ly*CDIM, nullptr, kb);
        k_gemm<EP_VT, CDIM, CDIM><<<gN1, tb256, 0, stream>>>(h, wtv, bv + (size_t)ly*CDIM, nullptr, vtb);

        k_attn<<<1024, tb128, 0, stream>>>(qb, kb, vtb, yb);

        k_gemm<EP_RESID, CDIM, CDIM><<<gN1, tb256, 0, stream>>>(yb, wtp, bp + (size_t)ly*CDIM, x, nullptr);

        k_ln<<<MALL, tb256, 0, stream>>>(x, ln2g + (size_t)ly*CDIM, ln2b + (size_t)ly*CDIM, h);

        k_gemm<EP_GELU, FF, CDIM><<<gN4, tb256, 0, stream>>>(h, wt1, b1 + (size_t)ly*FF, nullptr, hid);
        k_gemm<EP_RESID, CDIM, FF><<<gN1, tb256, 0, stream>>>(hid, wt2, b2 + (size_t)ly*CDIM, x, nullptr);
    }

    k_head<<<BB * TT, tb256, 0, stream>>>(x, lnfg, lnfb, headw, out);
}
